// LSTM_37864431681628
// MI455X (gfx1250) — matrix operand order repaired and checked by reading, not compiled, not run on hardware
//
#include <hip/hip_runtime.h>
#include <hip/hip_bf16.h>

// ---------------------------------------------------------------------------
// 4-layer LSTM (B=1024, T=128, H=200) fused persistent kernel for gfx1250.
//
// Latency-bound workload (weights ~2.5MB f16 -> L2-resident; HBM ~5MB total):
//  * 64 WGs x 256 threads (8 waves); each WG owns a 16-batch-row tile
//  * all recurrent state (h f16, c f32, z f32) in LDS, all 128 steps in-kernel
//  * GEMMs via v_wmma_f32_16x16x32_f16 (f16 in, f32 accumulate)
//  * gate dim padded 800 -> 4*208 = 832 (52 n-tiles split over 8 waves)
//  * K padded 200 -> 224; layers 2..4 use stacked [W_ih; W_hh] K=448
//  * bias folded into weight column K=223 (hbuf col 223 == 1.0 constant)
//  * layer-1 scalar input folded into weight column K=222 (hbuf0 col 222 = x_t)
//  * next timestep's x column prefetched via TDM tensor_load_to_lds
//    (2D strided tile), synchronized with s_wait_tensorcnt
// ---------------------------------------------------------------------------

typedef __attribute__((ext_vector_type(16))) _Float16     v16h;
typedef __attribute__((ext_vector_type(8)))  float        v8f;
typedef __attribute__((ext_vector_type(4)))  unsigned int v4u;
typedef __attribute__((ext_vector_type(4)))  int          i32x4;
typedef __attribute__((ext_vector_type(8)))  int          i32x8;
typedef __attribute__((ext_vector_type(4)))  unsigned int u32x4;

union U16 { v16h v; v4u q[2]; };

#define HID   200
#define SEQT  128
#define NTIL  52            // 4 gates * 13 tiles of 16 (padded 208 per gate)
#define ZROW  832           // padded gate row (4*208)
#define HROW  224           // padded K per matrix (halves per hbuf row)

// packed weight regions (f16 fragments, 1KB per 32x16 fragment)
#define W0_BYTES   (NTIL * 7  * 1024)   // layer0: rec-only, KC=7  -> 372736
#define WL_BYTES   (NTIL * 14 * 1024)   // layers1..3: KC=14       -> 745472
#define PACK_HALVES ((W0_BYTES + 3 * WL_BYTES) / 2)

// LDS partition (dynamic shared memory; extern smem starts at LDS offset 0)
#define HBUF_BYTES (4 * 16 * HROW * 2)          // 28672
#define CBUF_BYTES (4 * 16 * 208 * 4)           // 53248
#define ZBUF_BYTES (16 * ZROW * 4)              // 53248
#define XSTAGE_OFF (HBUF_BYTES + CBUF_BYTES + ZBUF_BYTES)  // 135168
#define SMEM_BYTES (XSTAGE_OFF + 64)            // + 16-float x staging buffer

#if defined(__has_builtin)
#if __has_builtin(__builtin_amdgcn_tensor_load_to_lds)
#define HAVE_TDM 1
#else
#define HAVE_TDM 0
#endif
#else
#define HAVE_TDM 0
#endif

// ---------------------------------------------------------------------------
// Weight packing: f32 sources -> f16 WMMA B-fragments in d_ws.
// Fragment layout: frag = [lane 0..31][16 halves], lane L supplies
// N = L&15, K = half + (L>=16 ? 16 : 0)  (contiguous k order, same as the A loads).
// ---------------------------------------------------------------------------
__global__ __launch_bounds__(256) void lstm_pack_w(
    const float* __restrict__ wih1, const float* __restrict__ whh1, const float* __restrict__ b1,
    const float* __restrict__ wih2, const float* __restrict__ whh2, const float* __restrict__ b2,
    const float* __restrict__ wih3, const float* __restrict__ whh3, const float* __restrict__ b3,
    const float* __restrict__ wih4, const float* __restrict__ whh4, const float* __restrict__ b4,
    _Float16* __restrict__ wpack)
{
    const int g = blockIdx.x * blockDim.x + threadIdx.x;
    if (g >= PACK_HALVES) return;

    const int frag   = g >> 9;          // 512 halves per fragment
    const int within = g & 511;
    const int lane   = within >> 4;
    const int hi     = within & 15;

    int l, fl, KC;
    if (frag < NTIL * 7) { l = 0; fl = frag; KC = 7; }
    else {
        const int f2 = frag - NTIL * 7;
        l  = 1 + f2 / (NTIL * 14);
        fl = f2 % (NTIL * 14);
        KC = 14;
    }
    const int nt = fl / KC;
    const int kc = fl % KC;

    const int Nl = lane & 15;
    const int Kl = hi + ((lane >> 4) << 4);
    const int n  = nt * 16 + Nl;        // padded gate index
    const int g4 = n / 208;             // gate (i,f,g,o)
    const int j  = n % 208;             // hidden unit within gate
    const int k  = kc * 32 + Kl;        // padded K index

    float v = 0.0f;
    if (j < HID) {
        const int row = g4 * HID + j;   // original [4H] row
        if (l == 0) {
            if      (k < HID)  v = whh1[row * HID + k];
            else if (k == 222) v = wih1[row];        // x column ([4H,1] weight)
            else if (k == 223) v = b1[row];          // bias column (h==1.0)
        } else {
            const float* wih = (l == 1) ? wih2 : (l == 2) ? wih3 : wih4;
            const float* whh = (l == 1) ? whh2 : (l == 2) ? whh3 : whh4;
            const float* bb  = (l == 1) ? b2   : (l == 2) ? b3   : b4;
            if (k < HROW) {                           // input section (h_{l-1})
                if (k < HID) v = wih[row * HID + k];
            } else {                                  // recurrent section (h_l)
                const int kk = k - HROW;
                if      (kk < HID)  v = whh[row * HID + kk];
                else if (kk == 223) v = bb[row];      // bias column
            }
        }
    }
    wpack[g] = (_Float16)v;
}

// ---------------------------------------------------------------------------
// TDM prefetch of x[:, t] column (16 rows, stride T floats) into LDS xstage.
// D# per CDNA5 ISA ch.8: group0 {count=1, lds_addr, global_addr, type=2},
// group1 {data_size=4B, tensor_dim0=1, tensor_dim1=16, tile_dim0=1,
//         tile_dim1=16, tensor_dim0_stride=T}.
// ---------------------------------------------------------------------------
__device__ __forceinline__ void tdm_prefetch_x(const float* gcol, unsigned ldsOff)
{
#if HAVE_TDM
    const unsigned long long ga = (unsigned long long)(const void*)gcol;
    u32x4 g0;
    g0[0] = 1u;                                           // count=1 (valid D#)
    g0[1] = ldsOff;                                       // lds_addr (bytes)
    g0[2] = (unsigned)(ga & 0xFFFFFFFFu);                 // global_addr[31:0]
    g0[3] = (unsigned)((ga >> 32) & 0x01FFFFFFu)          // global_addr[56:32]
          | 0x80000000u;                                  // type=2 ("image")
    i32x8 g1;
    g1[0] = 0x20000;      // data_size = 2 (4 bytes), no multicast/flags
    g1[1] = 1 << 16;      // tensor_dim0 = 1 (bits 79:48)
    g1[2] = 16 << 16;     // tensor_dim1 = 16 (bits 111:80)
    g1[3] = 1 << 16;      // tile_dim0 = 1 (bits 127:112)
    g1[4] = 16;           // tile_dim1 = 16 rows
    g1[5] = SEQT;         // tensor_dim0_stride = 128 elements (bits 207:160)
    g1[6] = 0;
    g1[7] = 0;
    const i32x4 z4 = {0, 0, 0, 0};
#if __has_include(<hip/amd_detail/amd_gfx1250_TDM.h>)
    const i32x8 z8 = {0, 0, 0, 0, 0, 0, 0, 0};
    __builtin_amdgcn_tensor_load_to_lds(g0, g1, z4, z4, z8, 0);   // clang-23 form
#else
    __builtin_amdgcn_tensor_load_to_lds(g0, g1, z4, z4, 0);       // ROCm 7.2 form
#endif
#else
    (void)gcol; (void)ldsOff;
#endif
}

__device__ __forceinline__ void wait_tensor0()
{
#if HAVE_TDM
#if defined(__has_builtin)
#if __has_builtin(__builtin_amdgcn_s_wait_tensorcnt)
    __builtin_amdgcn_s_wait_tensorcnt(0);
#endif
#endif
#endif
}

// ---------------------------------------------------------------------------
// Per-layer GEMM: z[16 x 832] = A[16 x 32*KC] * W[32*KC x 832]
// A assembled from LDS h rows (two ds_load_b128 per 32-K chunk),
// B streamed from packed weights (L2-resident), f32 accumulate in WMMA.
// ---------------------------------------------------------------------------
template<int KC>
__device__ __forceinline__ void gemm_layer(
    const _Float16* __restrict__ aSrc0,   // input-section h rows (LDS)
    const _Float16* __restrict__ aSrc1,   // recurrent-section h rows (LDS)
    const unsigned char* __restrict__ wbaseL,
    float* __restrict__ zbuf, int lane, int ntStart, int ntCnt)
{
    const int m16 = lane & 15;
    const int sel = lane >> 4;

    // A fragments for this layer (shared across this wave's n-tiles)
    v16h a[KC];
#pragma unroll
    for (int kc = 0; kc < KC; ++kc) {
        const _Float16* src = (KC == 14 && kc >= 7) ? aSrc1 : aSrc0;
        const int kk        = (KC == 14 && kc >= 7) ? (kc - 7) : kc;
        const unsigned char* rp =
            (const unsigned char*)(src + m16 * HROW) + kk * 64 + sel * 32;
        U16 u;
        u.q[0] = *(const v4u*)(rp);
        u.q[1] = *(const v4u*)(rp + 16);
        a[kc] = u.v;
    }

    const unsigned char* wl = wbaseL + lane * 32;
    for (int ni = 0; ni < ntCnt; ++ni) {
        const int nt = ntStart + ni;
        const unsigned char* fp = wl + nt * KC * 1024;
        __builtin_prefetch(fp + KC * 1024, 0, 0);     // next n-tile's fragments

        v8f acc = {0.f, 0.f, 0.f, 0.f, 0.f, 0.f, 0.f, 0.f};
#pragma unroll
        for (int kc = 0; kc < KC; ++kc) {
            U16 b;
            b.q[0] = *(const v4u*)(fp);
            b.q[1] = *(const v4u*)(fp + 16);
            fp += 1024;
            acc = __builtin_amdgcn_wmma_f32_16x16x32_f16(
                false, a[kc], false, b.v, (short)0, acc, false, false);
        }
        // D layout: VGPR r -> M = r + sel*8, lane -> N = lane&15
        float* zp = zbuf + (sel * 8) * ZROW + nt * 16 + m16;
#pragma unroll
        for (int r = 0; r < 8; ++r) zp[r * ZROW] = acc[r];
    }
}

// ---------------------------------------------------------------------------
// Persistent LSTM kernel: one WG = 16 batch rows, 256 threads = 8 waves.
// Waves 0..3 take 7 n-tiles each, waves 4..7 take 6 (total 52).
// ---------------------------------------------------------------------------
__global__ __launch_bounds__(256) void lstm_persist(
    const float* __restrict__ x,          // [B, T, 1]
    const _Float16* __restrict__ wpack,
    const float* __restrict__ fc_w,       // [1, 200]
    const float* __restrict__ fc_b,       // [1]
    float* __restrict__ out)              // [B, 1]
{
    extern __shared__ unsigned char smem[];
    _Float16* hbuf   = (_Float16*)smem;                          // [4][16][224] f16
    float*    cbuf   = (float*)(smem + HBUF_BYTES);              // [4][16][208] f32
    float*    zbuf   = (float*)(smem + HBUF_BYTES + CBUF_BYTES); // [16][832]   f32
    float*    xstage = (float*)(smem + XSTAGE_OFF);              // [16]        f32

    const int tid  = threadIdx.x;
    const int lane = tid & 31;
    const int wv   = tid >> 5;
    const int wgB  = blockIdx.x << 4;

    const int ntStart = (wv < 4) ? wv * 7 : 28 + (wv - 4) * 6;
    const int ntCnt   = (wv < 4) ? 7 : 6;

    // zero-init state; set constant-1 column (K=223) for bias folding
    for (int i = tid; i < 4 * 16 * HROW; i += 256) hbuf[i] = (_Float16)0.f;
    for (int i = tid; i < 4 * 16 * 208;  i += 256) cbuf[i] = 0.f;
    __syncthreads();
    if (tid < 64) hbuf[tid * HROW + 223] = (_Float16)1.f;
    __syncthreads();

    // kick off TDM fetch of x[:, 0]
    if (wv == 0) tdm_prefetch_x(x + (size_t)wgB * SEQT, XSTAGE_OFF);

    const unsigned char* wp = (const unsigned char*)wpack;

    for (int t = 0; t < SEQT; ++t) {
        wait_tensor0();            // wave0: x column has landed in LDS
        __syncthreads();
        // inject x_t into layer-0's K=222 column (weight column carries w_ih1)
#if HAVE_TDM
        if (tid < 16) hbuf[tid * HROW + 222] = (_Float16)xstage[tid];
#else
        if (tid < 16) hbuf[tid * HROW + 222] = (_Float16)x[(wgB + tid) * SEQT + t];
#endif
        __syncthreads();

#pragma unroll 1
        for (int l = 0; l < 4; ++l) {
            if (l == 0)
                gemm_layer<7>(hbuf, hbuf, wp, zbuf, lane, ntStart, ntCnt);
            else
                gemm_layer<14>(hbuf + (l - 1) * 16 * HROW, hbuf + l * 16 * HROW,
                               wp + W0_BYTES + (l - 1) * WL_BYTES,
                               zbuf, lane, ntStart, ntCnt);
            __syncthreads();

            // elementwise gate stage: c' = sig(f)*c + sig(i)*tanh(g),
            // h = sig(o)*tanh(c'); 16*208 units over 256 threads
            for (int u = tid; u < 16 * 208; u += 256) {
                const int m = u / 208;
                const int j = u - m * 208;
                if (j < HID) {
                    const float zi = zbuf[m * ZROW + j];
                    const float zf = zbuf[m * ZROW + 208 + j];
                    const float zg = zbuf[m * ZROW + 416 + j];
                    const float zo = zbuf[m * ZROW + 624 + j];
                    float c = cbuf[(l * 16 + m) * 208 + j];
                    const float si = 1.f / (1.f + __expf(-zi));
                    const float sf = 1.f / (1.f + __expf(-zf));
                    const float so = 1.f / (1.f + __expf(-zo));
                    c = sf * c + si * tanhf(zg);
                    cbuf[(l * 16 + m) * 208 + j] = c;
                    hbuf[(l * 16 + m) * HROW + j] = (_Float16)(so * tanhf(c));
                }
            }
            __syncthreads();
        }

        // overlap next timestep's x column fetch with nothing left to do this
        // step; xstage readers are past the barriers above
        if (wv == 0 && t + 1 < SEQT)
            tdm_prefetch_x(x + (size_t)wgB * SEQT + (t + 1), XSTAGE_OFF);
    }

    // final head: out = h4 @ fc_w.T + fc_b ; 16 threads per batch row
    {
        const int m  = tid >> 4;      // 0..15
        const int jj = tid & 15;
        float partial = 0.f;
        const _Float16* hr = hbuf + (48 + m) * HROW;
        for (int j = jj; j < HID; j += 16) partial += (float)hr[j] * fc_w[j];
        zbuf[m * 16 + jj] = partial;
        __syncthreads();
        if (tid < 16) {
            float s = fc_b[0];
            for (int k = 0; k < 16; ++k) s += zbuf[tid * 16 + k];
            out[wgB + tid] = s;
        }
    }
}

// ---------------------------------------------------------------------------
extern "C" void kernel_launch(void* const* d_in, const int* in_sizes, int n_in,
                              void* d_out, int out_size, void* d_ws, size_t ws_size,
                              hipStream_t stream) {
    const float* x    = (const float*)d_in[0];
    const float* wih1 = (const float*)d_in[1];
    const float* whh1 = (const float*)d_in[2];
    const float* b1   = (const float*)d_in[3];
    const float* wih2 = (const float*)d_in[4];
    const float* whh2 = (const float*)d_in[5];
    const float* b2   = (const float*)d_in[6];
    const float* wih3 = (const float*)d_in[7];
    const float* whh3 = (const float*)d_in[8];
    const float* b3   = (const float*)d_in[9];
    const float* wih4 = (const float*)d_in[10];
    const float* whh4 = (const float*)d_in[11];
    const float* b4   = (const float*)d_in[12];
    const float* fc_w = (const float*)d_in[13];
    const float* fc_b = (const float*)d_in[14];

    _Float16* wpack = (_Float16*)d_ws;   // ~2.55 MB packed f16 fragments

    const int packThreads = 256;
    const int packBlocks  = (PACK_HALVES + packThreads - 1) / packThreads;
    lstm_pack_w<<<packBlocks, packThreads, 0, stream>>>(
        wih1, whh1, b1, wih2, whh2, b2, wih3, whh3, b3, wih4, whh4, b4, wpack);

    lstm_persist<<<64, 256, SMEM_BYTES, stream>>>(
        x, wpack, fc_w, fc_b, (float*)d_out);
}
